// RnnOutputToEmbedding_84593675862137
// MI455X (gfx1250) — matrix operand order repaired and checked by reading, not compiled, not run on hardware
//
#include <hip/hip_runtime.h>

#define VOCAB  50257
#define HDIM   1024
#define NROWS  4096
#define NTILE  3142            // ceil(50257/16)
#define NPAIR  1571            // NTILE/2
#define KCH    32              // 1024 / 32
#define MBLK   32              // rows per workgroup
#define MT     2               // 16-row tiles per workgroup

typedef __attribute__((ext_vector_type(16))) __bf16 v16bf;
typedef __attribute__((ext_vector_type(8)))  float  v8f;

union FragU {
    unsigned int u[8];
    uint4        q[2];
    v16bf        b;
};

static __device__ __forceinline__ unsigned short f32_bf16_rn(float f) {
    unsigned int u = __float_as_uint(f);
    unsigned int r = u + 0x7FFFu + ((u >> 16) & 1u);
    return (unsigned short)(r >> 16);
}
static __device__ __forceinline__ float bf16_f32(unsigned short h) {
    return __uint_as_float(((unsigned int)h) << 16);
}

// ---------------------------------------------------------------------------
// Prep: var_w fp32 -> (hi, lo) bf16 fragments, pre-swizzled into the
// contiguous-k 16x16x32 B-fragment layout used here, same as A (lane = N, lane/16 = K half,
// dword d = K pair 2d).  One block per 16-row vocab tile.
// ---------------------------------------------------------------------------
__global__ __launch_bounds__(256) void prep_w_kernel(
    const float* __restrict__ w,
    unsigned int* __restrict__ whi,
    unsigned int* __restrict__ wlo)
{
    __shared__ unsigned short shi[16 * HDIM];
    __shared__ unsigned short slo[16 * HDIM];

    const int nt = blockIdx.x;
    const int t  = threadIdx.x;

    // Coalesced load of 16 vocab rows, split fp32 -> bf16 hi + bf16 lo.
    for (int j = 0; j < 64; ++j) {
        int idx = t + 256 * j;                 // 0 .. 16383
        int r   = idx >> 10;
        int c   = idx & (HDIM - 1);
        int row = nt * 16 + r;
        float x = (row < VOCAB) ? w[(size_t)row * HDIM + c] : 0.0f;
        unsigned short h = f32_bf16_rn(x);
        float lo = x - bf16_f32(h);
        shi[idx] = h;
        slo[idx] = f32_bf16_rn(lo);
    }
    __syncthreads();

    // Emit swizzled fragments: frag dword index == t (lane = t/8, d = t%8).
    const int lane = t >> 3;
    const int d    = t & 7;
    const int nl   = lane & 15;
    const int grp  = lane >> 4;
    for (int kc = 0; kc < KCH; ++kc) {
        int K0 = kc * 32 + 2 * d + grp * 16;
        unsigned int hi = (unsigned int)shi[nl * HDIM + K0] |
                          ((unsigned int)shi[nl * HDIM + K0 + 1] << 16);
        unsigned int lo = (unsigned int)slo[nl * HDIM + K0] |
                          ((unsigned int)slo[nl * HDIM + K0 + 1] << 16);
        size_t off = ((size_t)nt * KCH + kc) * 256 + t;
        whi[off] = hi;
        wlo[off] = lo;
    }
}

static __device__ __forceinline__ v16bf ld_frag(const unsigned int* p) {
    FragU f;
    const uint4* q = (const uint4*)p;
    f.q[0] = q[0];
    f.q[1] = q[1];
    return f.b;
}

// ---------------------------------------------------------------------------
// Main: bf16x3 WMMA GEMM + running argmax + gather/scale of embedding rows.
// One workgroup (8 waves) owns 32 rows x the whole vocab.
// ---------------------------------------------------------------------------
__global__ __launch_bounds__(256) void gemm_argmax_emb_kernel(
    const float* __restrict__ x,
    const unsigned int* __restrict__ whi,
    const unsigned int* __restrict__ wlo,
    const float* __restrict__ emb,
    float* __restrict__ out)
{
    // A fragments (hi/lo) for 2 M-tiles x 32 K-chunks: 128 KB
    __shared__ __align__(16) unsigned int a_sw[MT * 2 * KCH * 256];
    __shared__ float red_v[8][MT][16][16];
    __shared__ int   red_i[8][MT][16][16];
    __shared__ float fin_v[MBLK];
    __shared__ int   fin_i[MBLK];

    const int t    = threadIdx.x;
    const int wg   = blockIdx.x;
    const int lane = t & 31;
    const int w    = t >> 5;

    // ---- Stage 1: build swizzled A (hi/lo) in LDS -------------------------
    {
        const int l   = t >> 3;          // fragment lane
        const int d   = t & 7;           // fragment dword
        const int m   = l & 15;          // A layout: M = lane % 16
        const int grp = l >> 4;          // K-half select
        for (int mt = 0; mt < MT; ++mt) {
            size_t rowoff = (size_t)(wg * MBLK + mt * 16 + m) * HDIM;
            for (int kc = 0; kc < KCH; ++kc) {
                int K0 = kc * 32 + 2 * d + grp * 16;
                float x0 = x[rowoff + K0];
                float x1 = x[rowoff + K0 + 1];
                unsigned short h0 = f32_bf16_rn(x0);
                unsigned short h1 = f32_bf16_rn(x1);
                unsigned short l0 = f32_bf16_rn(x0 - bf16_f32(h0));
                unsigned short l1 = f32_bf16_rn(x1 - bf16_f32(h1));
                a_sw[((mt * 2 + 0) * KCH + kc) * 256 + t] =
                    (unsigned int)h0 | ((unsigned int)h1 << 16);
                a_sw[((mt * 2 + 1) * KCH + kc) * 256 + t] =
                    (unsigned int)l0 | ((unsigned int)l1 << 16);
            }
        }
    }
    __syncthreads();

    // ---- Stage 2: vocab sweep with WMMA, track per-lane argmax ------------
    float bv[MT][8];
    int   bi[MT][8];
#pragma unroll
    for (int mt = 0; mt < MT; ++mt)
#pragma unroll
        for (int i = 0; i < 8; ++i) { bv[mt][i] = -INFINITY; bi[mt][i] = 0; }

    for (int p = w; p < NPAIR; p += 8) {
        const int nt0 = 2 * p;
        const int nt1 = 2 * p + 1;
        v8f acc[MT][2];
#pragma unroll
        for (int mt = 0; mt < MT; ++mt) { acc[mt][0] = (v8f)0.0f; acc[mt][1] = (v8f)0.0f; }

        const unsigned int* bh0 = whi + ((size_t)nt0 * KCH) * 256 + lane * 8;
        const unsigned int* bl0 = wlo + ((size_t)nt0 * KCH) * 256 + lane * 8;
        const unsigned int* bh1 = whi + ((size_t)nt1 * KCH) * 256 + lane * 8;
        const unsigned int* bl1 = wlo + ((size_t)nt1 * KCH) * 256 + lane * 8;

        for (int kc = 0; kc < KCH; ++kc) {
            v16bf Bh0 = ld_frag(bh0 + kc * 256);
            v16bf Bl0 = ld_frag(bl0 + kc * 256);
            v16bf Bh1 = ld_frag(bh1 + kc * 256);
            v16bf Bl1 = ld_frag(bl1 + kc * 256);
#pragma unroll
            for (int mt = 0; mt < MT; ++mt) {
                v16bf Ah = ld_frag(&a_sw[((mt * 2 + 0) * KCH + kc) * 256 + lane * 8]);
                v16bf Al = ld_frag(&a_sw[((mt * 2 + 1) * KCH + kc) * 256 + lane * 8]);
                acc[mt][0] = __builtin_amdgcn_wmma_f32_16x16x32_bf16(
                    false, Al, false, Bh0, (short)0, acc[mt][0], false, false);
                acc[mt][0] = __builtin_amdgcn_wmma_f32_16x16x32_bf16(
                    false, Ah, false, Bl0, (short)0, acc[mt][0], false, false);
                acc[mt][0] = __builtin_amdgcn_wmma_f32_16x16x32_bf16(
                    false, Ah, false, Bh0, (short)0, acc[mt][0], false, false);
                acc[mt][1] = __builtin_amdgcn_wmma_f32_16x16x32_bf16(
                    false, Al, false, Bh1, (short)0, acc[mt][1], false, false);
                acc[mt][1] = __builtin_amdgcn_wmma_f32_16x16x32_bf16(
                    false, Ah, false, Bl1, (short)0, acc[mt][1], false, false);
                acc[mt][1] = __builtin_amdgcn_wmma_f32_16x16x32_bf16(
                    false, Ah, false, Bh1, (short)0, acc[mt][1], false, false);
            }
        }

        // Running argmax update (strict > keeps the lowest index on ties;
        // this lane's columns are visited in increasing order).
        const int c0 = nt0 * 16 + (lane & 15);
        const int c1 = c0 + 16;
#pragma unroll
        for (int mt = 0; mt < MT; ++mt) {
#pragma unroll
            for (int i = 0; i < 8; ++i) {
                float v0 = acc[mt][0][i];
                if (c0 < VOCAB && v0 > bv[mt][i]) { bv[mt][i] = v0; bi[mt][i] = c0; }
                float v1 = acc[mt][1][i];
                if (c1 < VOCAB && v1 > bv[mt][i]) { bv[mt][i] = v1; bi[mt][i] = c1; }
            }
        }
    }

    // ---- Stage 3: cross-lane / cross-wave argmax reduction ----------------
    {
        const int grp = lane >> 4;
        const int nl  = lane & 15;
#pragma unroll
        for (int mt = 0; mt < MT; ++mt)
#pragma unroll
            for (int i = 0; i < 8; ++i) {
                red_v[w][mt][grp * 8 + i][nl] = bv[mt][i];
                red_i[w][mt][grp * 8 + i][nl] = bi[mt][i];
            }
    }
    __syncthreads();

    if (t < MBLK) {
        const int mt = t >> 4;
        const int m  = t & 15;
        float best = -INFINITY;
        int   idx  = 0x7fffffff;
        for (int ww = 0; ww < 8; ++ww)
            for (int c = 0; c < 16; ++c) {
                float v = red_v[ww][mt][m][c];
                int  id = red_i[ww][mt][m][c];
                if (v > best || (v == best && id < idx)) { best = v; idx = id; }
            }
        fin_v[t] = best;
        fin_i[t] = idx;
    }
    __syncthreads();

    // ---- Stage 4: out[row] = max_logit * embedding[argmax] ----------------
    {
        const int r   = t >> 3;          // 8 threads per row
        const int sub = t & 7;
        float  val  = fin_v[r];
        size_t erow = (size_t)fin_i[r] * HDIM;
        size_t orow = (size_t)(wg * MBLK + r) * HDIM;
        const float4* ep = (const float4*)(emb + erow);
        float4*       op = (float4*)(out + orow);
        for (int j = 0; j < 32; ++j) {
            int c4 = sub + 8 * j;        // 0 .. 255
            float4 e = ep[c4];
            op[c4] = make_float4(val * e.x, val * e.y, val * e.z, val * e.w);
        }
    }
}

// ---------------------------------------------------------------------------
extern "C" void kernel_launch(void* const* d_in, const int* in_sizes, int n_in,
                              void* d_out, int out_size, void* d_ws, size_t ws_size,
                              hipStream_t stream)
{
    const float* x   = (const float*)d_in[0];   // [4096,1,1024] f32
    const float* vw  = (const float*)d_in[1];   // [50257,1024] f32
    const float* emb = (const float*)d_in[2];   // [50257,1024] f32
    float*       out = (float*)d_out;           // [4096,1,1024] f32

    unsigned int* whi = (unsigned int*)d_ws;                       // ~103 MB
    unsigned int* wlo = whi + (size_t)NTILE * KCH * 256;           // ~103 MB

    prep_w_kernel<<<NTILE, 256, 0, stream>>>(vw, whi, wlo);
    gemm_argmax_emb_kernel<<<NROWS / MBLK, 256, 0, stream>>>(x, whi, wlo, emb, out);
}
